// AMRLMax_18528488915178
// MI455X (gfx1250) — matrix operand order repaired and checked by reading, not compiled, not run on hardware
//
#include <hip/hip_runtime.h>
#include <math.h>
#include <stdint.h>

// ---------------- problem constants ----------------
#define B_  256
#define T_  256
#define I_  512
#define H_  512
#define G4  2048          // 4*H
#define HALF 256

typedef __bf16 bf16;
typedef __attribute__((ext_vector_type(16))) __bf16 v16bf;
typedef __attribute__((ext_vector_type(8)))  float  v8f;

// ---------------- WMMA fragment loaders (LDS -> VGPR, bf16 16x16x32 layouts) ----
// A-matrix 16x32 (MxK): lane L: m = m0 + (L&15), half = L>>4
//   element e (=2v+p): K = k0 + 16*(v>>2) + 8*half + 2*(v&3) + p
__device__ __forceinline__ v16bf fragA(const bf16* s, int m0, int k0, int ld, int lane) {
    const bf16* p = s + (size_t)(m0 + (lane & 15)) * ld;
    const int half = lane >> 4;
    v16bf r;
#pragma unroll
    for (int v = 0; v < 8; ++v) {
        int k = k0 + ((v >> 2) << 4) + (half << 3) + ((v & 3) << 1);
        r[2 * v]     = p[k];
        r[2 * v + 1] = p[k + 1];
    }
    return r;
}

// B-matrix 32x16 (KxN), stored in LDS as W[n][k] row-major:
//   lane L: n = n0 + (L&15); element e (=2v+p): K = k0 + 16*(v>>2) + 8*(L>>4) + 2*(v&3) + p
__device__ __forceinline__ v16bf fragB(const bf16* s, int n0, int k0, int ld, int lane) {
    const bf16* p = s + (size_t)(n0 + (lane & 15)) * ld;
    const int kb = k0 + ((lane >> 4) << 3);
    v16bf r;
#pragma unroll
    for (int v = 0; v < 8; ++v) {
        r[2 * v]     = p[kb + ((v >> 2) << 4) + ((v & 3) << 1)];
        r[2 * v + 1] = p[kb + ((v >> 2) << 4) + ((v & 3) << 1) + 1];
    }
    return r;
}

__device__ __forceinline__ v8f wmma_bf16(v16bf a, v16bf b, v8f c) {
    return __builtin_amdgcn_wmma_f32_16x16x32_bf16(false, a, false, b, (short)0, c,
                                                   false, false);
}

// ---------------- CDNA5 async copy: global -> LDS, 16B per lane --------------
// VGLOBAL GLOBAL_LOAD_ASYNC_TO_LDS_B128 (GVS mode: SADDR base + 32-bit VGPR off)
__device__ __forceinline__ void async_copy16(unsigned lds_addr, const void* sbase,
                                             unsigned voff) {
    asm volatile("global_load_async_to_lds_b128 %0, %1, %2"
                 :: "v"(lds_addr), "v"(voff), "s"(sbase)
                 : "memory");
}
__device__ __forceinline__ void wait_async0() {
    asm volatile("s_wait_asynccnt 0x0" ::: "memory");
}
// flat LDS pointer -> 32-bit LDS byte address (ISA 10.2: LDS offset = addr[31:0])
__device__ __forceinline__ unsigned lds_addr_of(const void* p) {
    return (unsigned)(uintptr_t)p;
}

// ---------------- prep: bf16 weight conversion + state init + barrier reset ----
__global__ __launch_bounds__(256)
void prep_kernel(const float* __restrict__ h0, const float* __restrict__ c0,
                 const float* __restrict__ Whh,
                 bf16* __restrict__ whh_bf, bf16* __restrict__ hbuf,
                 float* __restrict__ cst, float* __restrict__ mst,
                 unsigned* __restrict__ sync) {
    const int idx    = blockIdx.x * blockDim.x + threadIdx.x;
    const int stride = gridDim.x * blockDim.x;
    for (int i = idx; i < G4 * H_; i += stride) whh_bf[i] = (bf16)Whh[i];
    for (int i = idx; i < B_ * H_; i += stride) {
        hbuf[i]           = (bf16)h0[i];
        hbuf[B_ * H_ + i] = (bf16)0.0f;
        cst[i]            = c0[i];
    }
    for (int i = idx; i < B_ * HALF; i += stride) mst[i] = -__builtin_inff();
    if (idx < 2) sync[idx] = 0u;
}

// ---------------- x_proj GEMM: [65536 x 2048 x 512], fp32 -> bf16 on the fly ----
// block tile 128x128, K-chunk 64, 8 waves; wave tile 64x32 (4x2 wmma tiles)
__global__ __launch_bounds__(256)
void xproj_gemm(const float* __restrict__ x, const float* __restrict__ Wih,
                const float* __restrict__ bih, const float* __restrict__ bhh,
                float* __restrict__ xp) {
    const int BM = 128, BN = 128, BK = 64;
    __shared__ __align__(16) bf16 As[BM * BK];
    __shared__ __align__(16) bf16 Bs[BN * BK];
    const int tid  = threadIdx.x;
    const int lane = tid & 31;
    const int wave = tid >> 5;
    const int r0 = blockIdx.x * BM;       // row in flattened [B*T]
    const int n0 = blockIdx.y * BN;       // col in [4H]
    const int mbase = 64 * (wave >> 2);
    const int nbase = 32 * (wave & 3);

    v8f acc[4][2];
    const v8f vz = {0.f, 0.f, 0.f, 0.f, 0.f, 0.f, 0.f, 0.f};
#pragma unroll
    for (int i = 0; i < 4; ++i)
#pragma unroll
        for (int j = 0; j < 2; ++j) acc[i][j] = vz;

    for (int kk = 0; kk < I_; kk += BK) {
        for (int i = tid; i < (BM * BK) / 4; i += 256) {
            int row = i >> 4;              // BK/4 = 16 float4 per row
            int c4  = (i & 15) << 2;
            float4 v = *(const float4*)(x + (size_t)(r0 + row) * I_ + kk + c4);
            bf16* d = &As[row * BK + c4];
            d[0] = (bf16)v.x; d[1] = (bf16)v.y; d[2] = (bf16)v.z; d[3] = (bf16)v.w;
        }
        for (int i = tid; i < (BN * BK) / 4; i += 256) {
            int row = i >> 4;
            int c4  = (i & 15) << 2;
            float4 v = *(const float4*)(Wih + (size_t)(n0 + row) * I_ + kk + c4);
            bf16* d = &Bs[row * BK + c4];
            d[0] = (bf16)v.x; d[1] = (bf16)v.y; d[2] = (bf16)v.z; d[3] = (bf16)v.w;
        }
        // prefetch next K-chunk sources (-> global_prefetch_b8)
        if (kk + BK < I_) {
            __builtin_prefetch(x   + (size_t)(r0 + (tid >> 1)) * I_ + kk + BK, 0, 1);
            __builtin_prefetch(Wih + (size_t)(n0 + (tid >> 1)) * I_ + kk + BK, 0, 1);
        }
        __syncthreads();
#pragma unroll
        for (int ks = 0; ks < BK; ks += 32) {
            v16bf a[4], b[2];
#pragma unroll
            for (int mi = 0; mi < 4; ++mi) a[mi] = fragA(As, mbase + 16 * mi, ks, BK, lane);
#pragma unroll
            for (int ni = 0; ni < 2; ++ni) b[ni] = fragB(Bs, nbase + 16 * ni, ks, BK, lane);
#pragma unroll
            for (int mi = 0; mi < 4; ++mi)
#pragma unroll
                for (int ni = 0; ni < 2; ++ni)
                    acc[mi][ni] = wmma_bf16(a[mi], b[ni], acc[mi][ni]);
        }
        __syncthreads();
    }

    const int hi = lane >> 4, cc = lane & 15;
#pragma unroll
    for (int mi = 0; mi < 4; ++mi)
#pragma unroll
        for (int ni = 0; ni < 2; ++ni) {
            int ng = n0 + nbase + 16 * ni + cc;
            float bias = bih[ng] + bhh[ng];
#pragma unroll
            for (int j = 0; j < 8; ++j) {
                int rg = r0 + mbase + 16 * mi + 8 * hi + j;   // rg = b*T + t
                int t = rg & (T_ - 1);
                int b = rg >> 8;
                xp[(size_t)t * (B_ * G4) + (size_t)b * G4 + ng] = acc[mi][ni][j] + bias;
            }
        }
}

// ---------------- device-wide barrier (persistent-kernel grid sync) ----------
__device__ __forceinline__ void grid_sync(unsigned* cnt, unsigned* gen, unsigned nblk) {
    __threadfence();          // publish this thread's global writes
    __syncthreads();
    if (threadIdx.x == 0) {
        unsigned g = __hip_atomic_load(gen, __ATOMIC_RELAXED, __HIP_MEMORY_SCOPE_AGENT);
        unsigned arrived =
            __hip_atomic_fetch_add(cnt, 1u, __ATOMIC_ACQ_REL, __HIP_MEMORY_SCOPE_AGENT) + 1u;
        if (arrived == nblk) {
            __hip_atomic_store(cnt, 0u, __ATOMIC_RELAXED, __HIP_MEMORY_SCOPE_AGENT);
            __hip_atomic_store(gen, g + 1u, __ATOMIC_RELEASE, __HIP_MEMORY_SCOPE_AGENT);
        } else {
            while (__hip_atomic_load(gen, __ATOMIC_ACQUIRE, __HIP_MEMORY_SCOPE_AGENT) == g)
                __builtin_amdgcn_s_sleep(2);
        }
    }
    __syncthreads();
}

// ---------------- persistent recurrent LSTM kernel --------------------------
// 32 blocks, each owns a 64(B-rows) x 64(H-cols) tile -> computes i/f/g/o tiles
// via WMMA. K=512 in 16 chunks of 32, double-buffered LDS fed by
// GLOBAL_LOAD_ASYNC_TO_LDS_B128 (ASYNCcnt), then the cell update in-register.
#define KC 32
__global__ __launch_bounds__(256)
void lstm_rec(const float* __restrict__ xp, const bf16* __restrict__ whh,
              bf16* __restrict__ hbuf, float* __restrict__ cst,
              float* __restrict__ mst, float* __restrict__ out,
              unsigned* __restrict__ sync) {
    __shared__ __align__(16) bf16 As2[2][64 * KC];       // h tiles   (rows x K)
    __shared__ __align__(16) bf16 Bs2[2][4 * 64 * KC];   // W_hh tiles (4 gates x ncols x K)
    const int tid  = threadIdx.x;
    const int lane = tid & 31;
    const int wave = tid >> 5;
    const int brow = (blockIdx.x >> 3) * 64;   // 4 row-blocks
    const int hc0  = (blockIdx.x & 7) * 64;    // 8 col-blocks
    const int wrow = 32 * (wave >> 2);         // wave tile: 32 rows x 16 hcols x 4 gates
    const int wcol = 16 * (wave & 3);
    const int hi = lane >> 4, cc = lane & 15;
    const unsigned nblk = gridDim.x;
    const v8f vz = {0.f, 0.f, 0.f, 0.f, 0.f, 0.f, 0.f, 0.f};

    // per-thread async segment coordinates (16B segments)
    const int arow = tid >> 2, asc = tid & 3;            // h tile: 64 rows x 4 segs

    for (int t = 0; t < T_; ++t) {
        v8f acc[2][4];
#pragma unroll
        for (int mt = 0; mt < 2; ++mt)
#pragma unroll
            for (int g = 0; g < 4; ++g) acc[mt][g] = vz;

        const bf16* hsrc = hbuf + (size_t)(t & 1) * (B_ * H_);

        // issue chunk 0 (buffer 0)
        {
            async_copy16(lds_addr_of(&As2[0][arow * KC + asc * 8]), hsrc,
                         (unsigned)((brow + arow) * (H_ * 2) + asc * 16));
#pragma unroll
            for (int q = 0; q < 4; ++q) {
                int s = tid + 256 * q;
                int lin = s >> 2, sc = s & 3;
                int g = lin >> 6, row = lin & 63;
                async_copy16(lds_addr_of(&Bs2[0][(g * 64 + row) * KC + sc * 8]), whh,
                             (unsigned)((g * H_ + hc0 + row) * (H_ * 2) + sc * 16));
            }
        }

        for (int c = 0; c < H_ / KC; ++c) {
            wait_async0();            // my chunk-c async loads have landed in LDS
            __syncthreads();          // ... and everyone else's too
            if (c + 1 < H_ / KC) {    // overlap chunk c+1 loads with chunk c WMMAs
                const int kb = (c + 1) * KC * 2;   // byte offset along K
                bf16* Ab = As2[(c + 1) & 1];
                bf16* Bb = Bs2[(c + 1) & 1];
                async_copy16(lds_addr_of(&Ab[arow * KC + asc * 8]), hsrc,
                             (unsigned)((brow + arow) * (H_ * 2) + kb + asc * 16));
#pragma unroll
                for (int q = 0; q < 4; ++q) {
                    int s = tid + 256 * q;
                    int lin = s >> 2, sc = s & 3;
                    int g = lin >> 6, row = lin & 63;
                    async_copy16(lds_addr_of(&Bb[(g * 64 + row) * KC + sc * 8]), whh,
                                 (unsigned)((g * H_ + hc0 + row) * (H_ * 2) + kb + sc * 16));
                }
            }
            const bf16* Ab = As2[c & 1];
            const bf16* Bb = Bs2[c & 1];
            v16bf a[2], b[4];
            a[0] = fragA(Ab, wrow,      0, KC, lane);
            a[1] = fragA(Ab, wrow + 16, 0, KC, lane);
#pragma unroll
            for (int g = 0; g < 4; ++g)
                b[g] = fragB(Bb + g * 64 * KC, wcol, 0, KC, lane);
#pragma unroll
            for (int mt = 0; mt < 2; ++mt)
#pragma unroll
                for (int g = 0; g < 4; ++g)
                    acc[mt][g] = wmma_bf16(a[mt], b[g], acc[mt][g]);
        }

        // ---- LSTM cell, in-register: each lane owns (16 rows) x (1 hcol) ----
        const int hc = hc0 + wcol + cc;
        bf16* hdst = hbuf + (size_t)((t + 1) & 1) * (B_ * H_);
#pragma unroll
        for (int mt = 0; mt < 2; ++mt) {
#pragma unroll
            for (int j = 0; j < 8; ++j) {
                int gb = brow + wrow + 16 * mt + 8 * hi + j;
                size_t xb = (size_t)t * (B_ * G4) + (size_t)gb * G4;
                float gi = xp[xb + hc]           + acc[mt][0][j];
                float gf = xp[xb + H_ + hc]      + acc[mt][1][j];
                float gg = xp[xb + 2 * H_ + hc]  + acc[mt][2][j];
                float go = xp[xb + 3 * H_ + hc]  + acc[mt][3][j];
                float si = 1.0f / (1.0f + __expf(-gi));
                float sf = 1.0f / (1.0f + __expf(-gf));
                float so = 1.0f / (1.0f + __expf(-go));
                float tg = tanhf(gg);
                size_t sidx = (size_t)gb * H_ + hc;
                float cn = sf * cst[sidx] + si * tg;
                float hn = so * tanhf(cn);
                cst[sidx]  = cn;
                hdst[sidx] = (bf16)hn;
                if (hc < HALF) {
                    size_t midx = (size_t)gb * HALF + hc;
                    float mo = fmaxf(mst[midx], hn);
                    mst[midx] = mo;
                    out[(size_t)gb * (T_ * HALF) + (size_t)t * HALF + hc] = mo;
                } else {
                    out[(size_t)(B_ + gb) * (T_ * HALF) + (size_t)t * HALF + (hc - HALF)] = hn;
                }
                if (t == T_ - 1) {
                    size_t ob = (size_t)2 * B_ * T_ * HALF;
                    out[ob + sidx]            = hn;   // h_n
                    out[ob + B_ * H_ + sidx]  = cn;   // c_n
                }
            }
        }
        grid_sync(sync, sync + 1, nblk);
    }
}

// ---------------- host launcher ---------------------------------------------
extern "C" void kernel_launch(void* const* d_in, const int* in_sizes, int n_in,
                              void* d_out, int out_size, void* d_ws, size_t ws_size,
                              hipStream_t stream) {
    const float* x   = (const float*)d_in[0];
    const float* h0  = (const float*)d_in[1];
    const float* c0  = (const float*)d_in[2];
    const float* Wih = (const float*)d_in[3];
    const float* Whh = (const float*)d_in[4];
    const float* bih = (const float*)d_in[5];
    const float* bhh = (const float*)d_in[6];
    float* out = (float*)d_out;

    char* ws = (char*)d_ws;
    const size_t XP_BYTES  = (size_t)T_ * B_ * G4 * sizeof(float);   // 512 MiB
    const size_t WHH_BYTES = (size_t)G4 * H_ * sizeof(bf16);         // 2 MiB
    const size_t HB_BYTES  = (size_t)2 * B_ * H_ * sizeof(bf16);     // 512 KiB
    const size_t C_BYTES   = (size_t)B_ * H_ * sizeof(float);        // 512 KiB
    const size_t M_BYTES   = (size_t)B_ * HALF * sizeof(float);      // 256 KiB
    float*    xp   = (float*)ws;
    bf16*     whhb = (bf16*)(ws + XP_BYTES);
    bf16*     hbuf = (bf16*)(ws + XP_BYTES + WHH_BYTES);
    float*    cst  = (float*)(ws + XP_BYTES + WHH_BYTES + HB_BYTES);
    float*    mst  = (float*)(ws + XP_BYTES + WHH_BYTES + HB_BYTES + C_BYTES);
    unsigned* sync = (unsigned*)(ws + XP_BYTES + WHH_BYTES + HB_BYTES + C_BYTES + M_BYTES);

    prep_kernel<<<1024, 256, 0, stream>>>(h0, c0, Whh, whhb, hbuf, cst, mst, sync);

    dim3 g1((B_ * T_) / 128, G4 / 128);  // 512 x 16 blocks
    xproj_gemm<<<g1, 256, 0, stream>>>(x, Wih, bih, bhh, xp);

    lstm_rec<<<32, 256, 0, stream>>>(xp, whhb, hbuf, cst, mst, out, sync);
}
